// MLPBlock_15427522527794
// MI455X (gfx1250) — matrix operand order repaired and checked by reading, not compiled, not run on hardware
//
#include <hip/hip_runtime.h>
#include <hip/hip_bf16.h>
#include <math.h>

// ---------------------------------------------------------------------------
// MI455X (gfx1250) fused GroupNorm + MLP block.
//   GEMM1: (16384 x 384) x (384 x 1536)  -> v_wmma_f32_16x16x32_f16
//   GEMM2: (16384 x 1536) x (1536 x 384) -> v_wmma_f32_16x16x32_f16
// ~39 GFLOP, ~110 MB traffic; intermediates L2-resident (192 MB L2).
// This revision: K-loop unrolled x2 with ping-pong operand sets (kills the
// v_dual_mov rotation chains + hazard v_nops), bank-conflict-free padded LDS
// panels (row stride % 64 words == 4), and async global->LDS staging of the
// shared panel in BOTH GEMMs (gfx1250 ASYNCcnt path).
// ---------------------------------------------------------------------------

#define BATCH    16
#define C_IN     384
#define C_EXP    1536
#define NPIX     1024      // 32*32 pixels per batch image
#define EMB_N    1024
#define NGROUPS  32
#define CPG      12        // channels per group = 384/32
#define GELEM    (CPG * NPIX)

#define LDP1     (C_IN + 8)      // 392 halves: gemm1 panel row stride (784 B)
#define KPH      768             // gemm2 K per staging phase
#define LDP2     (KPH + 8)       // 776 halves: gemm2 panel row stride (1552 B)

typedef __attribute__((ext_vector_type(8)))  _Float16 v8h;
typedef __attribute__((ext_vector_type(16))) _Float16 v16h;
typedef __attribute__((ext_vector_type(8)))  float    v8f;
typedef __attribute__((ext_vector_type(4)))  float    v4f;

// ---------------------------------------------------------------------------
// WMMA operand packing (ISA 7.12.2 VGPR layouts, wave32).
// A 16x32 f16 (MxK): lane hi*16+r holds row M=r; v16h[0..7]=K[k0+8hi..+7],
// v16h[8..15]=K[k0+16+8hi..+7] -> two 16B contiguous loads.
// ---------------------------------------------------------------------------
__device__ __forceinline__ v16h load_matA(const _Float16* S,
                                          int ld, int r, int hi, int k0) {
    const _Float16* p = S + (size_t)r * ld + k0 + 8 * hi;
    v8h lo = *(const v8h*)(p);
    v8h hh = *(const v8h*)(p + 16);
    return __builtin_shufflevector(lo, hh, 0,1,2,3,4,5,6,7,8,9,10,11,12,13,14,15);
}

// B 32x16 f16 (KxN): lane r (col N=r) in half hi holds K=k0+8hi..+7 and
// K=k0+16+8hi..+7 -> two 16B loads from row-major S where S[n][k] is column n.
__device__ __forceinline__ v16h load_matB(const _Float16* S,
                                          int ld, int r, int hi, int k0) {
    const _Float16* p = S + (size_t)r * ld + k0 + 8 * hi;
    v8h lo = *(const v8h*)(p);
    v8h hh = *(const v8h*)(p + 16);
    return __builtin_shufflevector(lo, hh, 0,1,2,3,4,5,6,7,8,9,10,11,12,13,14,15);
}

__device__ __forceinline__ v8f wmma_f16(v16h a, v16h b, v8f c) {
    return __builtin_amdgcn_wmma_f32_16x16x32_f16(false, a, false, b,
                                                  (short)0, c, false, false);
}

__device__ __forceinline__ void async_lds_b128(unsigned laddr,
                                               const void* gptr) {
    unsigned long long ga = (unsigned long long)(size_t)gptr;
    asm volatile("global_load_async_to_lds_b128 %0, %1, off"
                 :: "v"(laddr), "v"(ga) : "memory");
}
__device__ __forceinline__ void wait_async0() {
    asm volatile("s_wait_asynccnt 0x0" ::: "memory");
}

// ---------------------------------------------------------------------------
// Kernel 0: fp32 -> f16 weight conversion.
// ---------------------------------------------------------------------------
__global__ void cvt_f16_kernel(const float* __restrict__ src,
                               _Float16* __restrict__ dst, int n) {
    int i = blockIdx.x * blockDim.x + threadIdx.x;
    if (i < n) dst[i] = (_Float16)src[i];
}

// ---------------------------------------------------------------------------
// Kernel 1: emb GEMM -> (1+scale) and shift tables.
// ---------------------------------------------------------------------------
__global__ void emb_mod_kernel(const float* __restrict__ emb,
                               const float* __restrict__ we,
                               const float* __restrict__ be,
                               float* __restrict__ scale1p,
                               float* __restrict__ shiftv) {
    const int j = blockIdx.x * blockDim.x + threadIdx.x;   // 0..3071
    const int b = blockIdx.y;
    const v4f* e = (const v4f*)(emb + (size_t)b * EMB_N);
    const v4f* w = (const v4f*)(we + (size_t)j * EMB_N);
    float acc = 0.f;
    #pragma unroll 4
    for (int k = 0; k < EMB_N / 4; ++k) {
        v4f ev = e[k], wv = w[k];
        acc += ev[0]*wv[0] + ev[1]*wv[1] + ev[2]*wv[2] + ev[3]*wv[3];
    }
    float val = acc + be[j];
    if (j < C_EXP) scale1p[b * C_EXP + j] = 1.0f + val;
    else           shiftv[b * C_EXP + (j - C_EXP)] = val;
}

// ---------------------------------------------------------------------------
// Kernel 2: GroupNorm(32) fused with transpose to (b, pixel, channel) f16.
// ---------------------------------------------------------------------------
__global__ void groupnorm_t_kernel(const float* __restrict__ x,
                                   const float* __restrict__ gw,
                                   const float* __restrict__ gb,
                                   _Float16* __restrict__ xnT) {
    __shared__ float buf[GELEM];     // 48 KB
    __shared__ float r0[256];
    __shared__ float r1[256];
    const int tid = threadIdx.x;
    const int b = blockIdx.x / NGROUPS;
    const int g = blockIdx.x % NGROUPS;
    const float* src = x + ((size_t)b * C_IN + g * CPG) * NPIX;

    float s = 0.f, ss = 0.f;
    for (int i = tid; i < GELEM; i += 256) {
        float v = src[i];
        buf[i] = v;
        s += v;
        ss += v * v;
    }
    r0[tid] = s; r1[tid] = ss;
    __syncthreads();
    for (int st = 128; st > 0; st >>= 1) {
        if (tid < st) { r0[tid] += r0[tid + st]; r1[tid] += r1[tid + st]; }
        __syncthreads();
    }
    const float inv_n = 1.0f / (float)GELEM;
    const float mu = r0[0] * inv_n;
    const float var = r1[0] * inv_n - mu * mu;
    const float rstd = rsqrtf(var + 1e-5f);

    for (int i = tid; i < GELEM; i += 256) {
        int ch = i >> 10;
        int p  = i & (NPIX - 1);
        int c  = g * CPG + ch;
        float v = (buf[i] - mu) * rstd * gw[c] + gb[c];
        xnT[((size_t)b * NPIX + p) * C_IN + c] = (_Float16)v;
    }
}

// ---------------------------------------------------------------------------
// Kernel 3: GEMM1 fused: h = silu(xn @ w1^T + b1) * (1+scale) + shift  (f16)
//   D[M=o][N=pixel].  Wave: 32 pixels x 64 channels = 8 acc tiles.
//   8 waves/block share one 32-pixel activation panel async-staged into a
//   bank-conflict-free padded LDS buffer; w1 operands ping-pong from global.
// ---------------------------------------------------------------------------
__global__ void __launch_bounds__(256)
gemm1_fused_kernel(const _Float16* __restrict__ xnT,  // [B][NPIX][C_IN]
                   const _Float16* __restrict__ w1h,  // [C_EXP][C_IN]
                   const float* __restrict__ b1,
                   const float* __restrict__ scale1p, // [B][C_EXP]
                   const float* __restrict__ shiftv,  // [B][C_EXP]
                   _Float16* __restrict__ hT) {       // [B][NPIX][C_EXP]
    __shared__ _Float16 panel[32 * LDP1];             // 25,088 B, padded rows
    const int tid  = threadIdx.x;
    const int lane = tid & 31;
    const int r = lane & 15, hi = lane >> 4;
    int wid = (blockIdx.x * blockDim.x + tid) >> 5;
    const int og  = wid % 24; wid /= 24;   // 24 strips of 64 channels
    const int pt2 = wid % 32;              // 32-pixel tiles
    const int b   = wid / 32;
    const int obase = og * 64;

    // ---- async stage: 32 rows x 384 halves into padded panel ----
    {
        const _Float16* src = xnT + ((size_t)b * NPIX + (size_t)pt2 * 32) * C_IN;
        unsigned lbase = (unsigned)(size_t)(&panel[0]);
        #pragma unroll
        for (int i = 0; i < 6; ++i) {                  // 32*48 chunks / 256
            int idx = tid + i * 256;
            int row = idx / 48, col = idx % 48;        // 48 x 16B per row
            async_lds_b128(lbase + (unsigned)(row * (LDP1 * 2) + col * 16),
                           src + (size_t)row * C_IN + (size_t)col * 8);
        }
        wait_async0();
        __syncthreads();
    }

    const _Float16* Ap = w1h + (size_t)obase * C_IN;
    v8f acc[2][4] = {};
    v16h am0[4], am1[4];
    #pragma unroll
    for (int t = 0; t < 4; ++t)
        am0[t] = load_matA(Ap + (size_t)(t * 16) * C_IN, C_IN, r, hi, 0);

    for (int k0 = 0; k0 < C_IN; k0 += 64) {            // 6 iters, 2 K-steps
        const int k1 = k0 + 32;
        #pragma unroll
        for (int t = 0; t < 4; ++t)
            am1[t] = load_matA(Ap + (size_t)(t * 16) * C_IN, C_IN, r, hi, k1);
        {
            v16h bm0 = load_matB(panel,             LDP1, r, hi, k0);
            v16h bm1 = load_matB(panel + 16 * LDP1, LDP1, r, hi, k0);
            #pragma unroll
            for (int t = 0; t < 4; ++t) {
                acc[0][t] = wmma_f16(am0[t], bm0, acc[0][t]);
                acc[1][t] = wmma_f16(am0[t], bm1, acc[1][t]);
            }
        }
        const int k2 = (k0 + 64 < C_IN) ? k0 + 64 : 0; // clamped prefetch
        #pragma unroll
        for (int t = 0; t < 4; ++t)
            am0[t] = load_matA(Ap + (size_t)(t * 16) * C_IN, C_IN, r, hi, k2);
        {
            v16h bm0 = load_matB(panel,             LDP1, r, hi, k1);
            v16h bm1 = load_matB(panel + 16 * LDP1, LDP1, r, hi, k1);
            #pragma unroll
            for (int t = 0; t < 4; ++t) {
                acc[0][t] = wmma_f16(am1[t], bm0, acc[0][t]);
                acc[1][t] = wmma_f16(am1[t], bm1, acc[1][t]);
            }
        }
    }

    const float* scrow = scale1p + (size_t)b * C_EXP;
    const float* shrow = shiftv  + (size_t)b * C_EXP;
    #pragma unroll
    for (int tp = 0; tp < 2; ++tp) {
        const int pix = pt2 * 32 + tp * 16 + r;
        _Float16* orow = hT + ((size_t)b * NPIX + pix) * C_EXP;
        #pragma unroll
        for (int t = 0; t < 4; ++t) {
            const int o0 = obase + t * 16 + 8 * hi;   // 8 consecutive channels
            v4f bia = *(const v4f*)(b1 + o0);
            v4f bib = *(const v4f*)(b1 + o0 + 4);
            v4f sca = *(const v4f*)(scrow + o0);
            v4f scb = *(const v4f*)(scrow + o0 + 4);
            v4f sha = *(const v4f*)(shrow + o0);
            v4f shb = *(const v4f*)(shrow + o0 + 4);
            v8h hv;
            #pragma unroll
            for (int v = 0; v < 8; ++v) {
                float bias = (v < 4) ? bia[v] : bib[v - 4];
                float sc   = (v < 4) ? sca[v] : scb[v - 4];
                float sh   = (v < 4) ? sha[v] : shb[v - 4];
                float val  = acc[tp][t][v] + bias;
                float sig  = 1.0f / (1.0f + __expf(-val));
                hv[v] = (_Float16)((val * sig) * sc + sh);
            }
            *(v8h*)(orow + o0) = hv;
        }
    }
}

// ---------------------------------------------------------------------------
// Kernel 4: GEMM2 fused residual: out = x + h @ w2^T + b2  (fp32 out)
//   D[M=pixel][N=j].  8 waves/block share one 32-pixel hT panel, async-staged
//   into padded LDS in two 48KB K-phases; each wave owns 48 channels
//   (3 j-tiles, 6 acc tiles).  w2 operands ping-pong from global.
// ---------------------------------------------------------------------------
__global__ void __launch_bounds__(256)
gemm2_fused_kernel(const _Float16* __restrict__ hT,   // [B][NPIX][C_EXP]
                   const _Float16* __restrict__ w2h,  // [C_IN][C_EXP]
                   const float* __restrict__ b2,
                   const float* __restrict__ x,
                   float* __restrict__ out) {
    __shared__ _Float16 panel[32 * LDP2];             // 49,664 B, padded rows
    const int tid  = threadIdx.x;
    const int lane = tid & 31;
    const int r = lane & 15, hi = lane >> 4;
    int wid = (blockIdx.x * blockDim.x + tid) >> 5;
    const int jg  = wid & 7; wid >>= 3;    // 8 strips of 48 out channels
    const int pt2 = wid % 32;              // 32-pixel tiles
    const int b   = wid / 32;

    const _Float16* srcA = hT + ((size_t)b * NPIX + (size_t)pt2 * 32) * C_EXP;
    const _Float16* Bp = w2h + (size_t)(jg * 48) * C_EXP;
    const unsigned lbase = (unsigned)(size_t)(&panel[0]);

    v8f acc[2][3] = {};
    for (int ph = 0; ph < 2; ++ph) {
        const int kbase = ph * KPH;
        __syncthreads();   // previous-phase LDS reads complete before refill
        #pragma unroll
        for (int i = 0; i < 12; ++i) {                 // 32*96 chunks / 256
            int idx = tid + i * 256;
            int row = idx / 96, col = idx % 96;        // 96 x 16B per row
            async_lds_b128(lbase + (unsigned)(row * (LDP2 * 2) + col * 16),
                           srcA + (size_t)row * C_EXP + kbase + (size_t)col * 8);
        }
        wait_async0();
        __syncthreads();

        v16h bm0[3], bm1[3];
        #pragma unroll
        for (int t = 0; t < 3; ++t)
            bm0[t] = load_matB(Bp + (size_t)(t * 16) * C_EXP, C_EXP, r, hi, kbase);

        for (int kk = 0; kk < KPH; kk += 64) {         // 12 iters, 2 K-steps
            const int k1 = kbase + kk + 32;
            #pragma unroll
            for (int t = 0; t < 3; ++t)
                bm1[t] = load_matB(Bp + (size_t)(t * 16) * C_EXP, C_EXP, r, hi, k1);
            {
                v16h a0 = load_matA(panel,             LDP2, r, hi, kk);
                v16h a1 = load_matA(panel + 16 * LDP2, LDP2, r, hi, kk);
                #pragma unroll
                for (int t = 0; t < 3; ++t) {
                    acc[0][t] = wmma_f16(a0, bm0[t], acc[0][t]);
                    acc[1][t] = wmma_f16(a1, bm0[t], acc[1][t]);
                }
            }
            const int k2 = kbase + ((kk + 64 < KPH) ? kk + 64 : 0);
            #pragma unroll
            for (int t = 0; t < 3; ++t)
                bm0[t] = load_matB(Bp + (size_t)(t * 16) * C_EXP, C_EXP, r, hi, k2);
            {
                v16h a0 = load_matA(panel,             LDP2, r, hi, kk + 32);
                v16h a1 = load_matA(panel + 16 * LDP2, LDP2, r, hi, kk + 32);
                #pragma unroll
                for (int t = 0; t < 3; ++t) {
                    acc[0][t] = wmma_f16(a0, bm1[t], acc[0][t]);
                    acc[1][t] = wmma_f16(a1, bm1[t], acc[1][t]);
                }
            }
        }
    }

    #pragma unroll
    for (int tp = 0; tp < 2; ++tp) {
        #pragma unroll
        for (int t = 0; t < 3; ++t) {
            const int j = jg * 48 + t * 16 + r;
            const float bj = b2[j];
            const size_t base = (((size_t)b * C_IN + j) << 10)
                              + (size_t)(pt2 * 32 + tp * 16 + 8 * hi);
            v4f x0 = *(const v4f*)(x + base);
            v4f x1 = *(const v4f*)(x + base + 4);
            v4f o0, o1;
            #pragma unroll
            for (int v = 0; v < 4; ++v) {
                o0[v] = x0[v] + acc[tp][t][v]     + bj;
                o1[v] = x1[v] + acc[tp][t][v + 4] + bj;
            }
            *(v4f*)(out + base)     = o0;
            *(v4f*)(out + base + 4) = o1;
        }
    }
}

// ---------------------------------------------------------------------------
// Host launcher.  Workspace layout (bytes):
//   xnT  f16 [16][1024][384]  @ 0          (12,582,912)
//   w1h  f16 [1536][384]      @ 12,582,912 ( 1,179,648)
//   w2h  f16 [384][1536]      @ 13,762,560 ( 1,179,648)
//   scale1p f32 [16][1536]    @ 14,942,208 (    98,304)
//   shift   f32 [16][1536]    @ 15,040,512 (    98,304)
//   hT   f16 [16][1024][1536] @ 15,138,816 (50,331,648)   total ~65.5 MB
// ---------------------------------------------------------------------------
extern "C" void kernel_launch(void* const* d_in, const int* in_sizes, int n_in,
                              void* d_out, int out_size, void* d_ws, size_t ws_size,
                              hipStream_t stream) {
    const float* x    = (const float*)d_in[0];
    const float* emb  = (const float*)d_in[1];
    const float* gn_w = (const float*)d_in[2];
    const float* gn_b = (const float*)d_in[3];
    const float* w1   = (const float*)d_in[4];
    const float* b1   = (const float*)d_in[5];
    const float* we   = (const float*)d_in[6];
    const float* be   = (const float*)d_in[7];
    const float* w2   = (const float*)d_in[8];
    const float* b2   = (const float*)d_in[9];
    float* out = (float*)d_out;

    char* ws = (char*)d_ws;
    _Float16* xnT     = (_Float16*)(ws);
    _Float16* w1h     = (_Float16*)(ws + 12582912);
    _Float16* w2h     = (_Float16*)(ws + 13762560);
    float*    scale1p = (float*)   (ws + 14942208);
    float*    shiftv  = (float*)   (ws + 15040512);
    _Float16* hT      = (_Float16*)(ws + 15138816);

    const int NW = C_EXP * C_IN;  // 589824 weights each
    cvt_f16_kernel<<<(NW + 255) / 256, 256, 0, stream>>>(w1, w1h, NW);
    cvt_f16_kernel<<<(NW + 255) / 256, 256, 0, stream>>>(w2, w2h, NW);

    emb_mod_kernel<<<dim3(2 * C_EXP / 256, BATCH), 256, 0, stream>>>(
        emb, we, be, scale1p, shiftv);

    groupnorm_t_kernel<<<BATCH * NGROUPS, 256, 0, stream>>>(x, gn_w, gn_b, xnT);

    // 16 batches * 32 pixel tiles * 24 strips = 12288 waves, 8/block
    gemm1_fused_kernel<<<1536, 256, 0, stream>>>(xnT, w1h, b1, scale1p, shiftv, hT);

    // 16 batches * 32 pixel tiles * 8 strips = 4096 waves, 8/block
    gemm2_fused_kernel<<<512, 256, 0, stream>>>(hT, w2h, b2, x, out);
}
